// SingleStreamBlock_37117107372193
// MI455X (gfx1250) — matrix operand order repaired and checked by reading, not compiled, not run on hardware
//
#include <hip/hip_runtime.h>
#include <hip/hip_bf16.h>

// ---------------- types ----------------
typedef __bf16 bf16;
typedef __bf16 bf16x16 __attribute__((ext_vector_type(16)));
typedef float  f32x8   __attribute__((ext_vector_type(8)));
typedef unsigned int u32x4 __attribute__((ext_vector_type(4)));

union FragB { bf16x16 v; u32x4 q[2]; };

#define HIDDEN 3072
#define SEQ    2048
#define NHEADS 24
#define HDIM   128
#define MLPH   12288
#define N1     (3*HIDDEN + MLPH)   // 21504
#define CATW   (HIDDEN + MLPH)     // 15360

__device__ __forceinline__ bf16 f2bf(float f) {
    unsigned u = __float_as_uint(f);
    unsigned r = u + 0x7FFFu + ((u >> 16) & 1u);   // round-to-nearest-even
    unsigned short s = (unsigned short)(r >> 16);
    return __builtin_bit_cast(bf16, s);
}

// ---------------- tiny kernels ----------------
__global__ void silu_k(const float* __restrict__ v, float* __restrict__ sv) {
    int i = blockIdx.x * 256 + threadIdx.x;
    if (i < HIDDEN) { float x = v[i]; sv[i] = x / (1.f + __expf(-x)); }
}

__global__ void mod_gemv(const float* __restrict__ sv, const float* __restrict__ W,
                         const float* __restrict__ b, float* __restrict__ mod) {
    int c = blockIdx.x * 256 + threadIdx.x;   // 9216 outputs
    float acc = b[c];
    for (int i = 0; i < HIDDEN; ++i)
        acc = fmaf(sv[i], W[(long)i * (3*HIDDEN) + c], acc);
    mod[c] = acc;
}

// fp32 W[K][N] -> bf16 Wt[N][K], 32x32 LDS tiles, coalesced both sides.
__global__ __launch_bounds__(256)
void cast_transpose(const float* __restrict__ src, bf16* __restrict__ dst, int K, int N) {
    __shared__ float tile[32][33];
    int kb = blockIdx.x * 32, nb = blockIdx.y * 32;
    int c = threadIdx.x & 31, r = threadIdx.x >> 5;   // 8 rows per pass
#pragma unroll
    for (int i = 0; i < 4; ++i)
        tile[r + 8*i][c] = src[(long)(kb + r + 8*i) * N + nb + c];
    __syncthreads();
#pragma unroll
    for (int i = 0; i < 4; ++i)
        dst[(long)(nb + r + 8*i) * K + kb + c] = f2bf(tile[c][r + 8*i]);
}

// bf16 V[h][pos][128] -> Vt[h][128][pos], 32x33 LDS tiles.
__global__ __launch_bounds__(256)
void transpose_v(const bf16* __restrict__ src, bf16* __restrict__ dst) {
    __shared__ bf16 tile[32][33];
    int h = blockIdx.z;
    int pb = blockIdx.x * 32, db = blockIdx.y * 32;
    int c = threadIdx.x & 31, r = threadIdx.x >> 5;
    const bf16* s = src + (long)h * SEQ * HDIM;
    bf16*       d = dst + (long)h * SEQ * HDIM;
#pragma unroll
    for (int i = 0; i < 4; ++i)
        tile[r + 8*i][c] = s[(long)(pb + r + 8*i) * HDIM + db + c];
    __syncthreads();
#pragma unroll
    for (int i = 0; i < 4; ++i)
        d[(long)(db + r + 8*i) * SEQ + pb + c] = tile[c][r + 8*i];
}

// LayerNorm + (1+scale)*xn + shift, cast to bf16. one block per row.
__global__ __launch_bounds__(256)
void ln_mod(const float* __restrict__ x, const float* __restrict__ mod, bf16* __restrict__ xm) {
    __shared__ float red[256];
    int r = blockIdx.x, t = threadIdx.x;
    const float* xr = x + (long)r * HIDDEN;
    float loc[12]; float s = 0.f;
#pragma unroll
    for (int j = 0; j < 12; ++j) { loc[j] = xr[j*256 + t]; s += loc[j]; }
    red[t] = s; __syncthreads();
    for (int k = 128; k > 0; k >>= 1) { if (t < k) red[t] += red[t+k]; __syncthreads(); }
    float mu = red[0] / (float)HIDDEN;
    __syncthreads();
    float s2 = 0.f;
#pragma unroll
    for (int j = 0; j < 12; ++j) { float d = loc[j] - mu; s2 += d * d; }
    red[t] = s2; __syncthreads();
    for (int k = 128; k > 0; k >>= 1) { if (t < k) red[t] += red[t+k]; __syncthreads(); }
    float inv = rsqrtf(red[0] / (float)HIDDEN + 1e-6f);
    bf16* xo = xm + (long)r * HIDDEN;
#pragma unroll
    for (int j = 0; j < 12; ++j) {
        int c = j*256 + t;
        float xn = (loc[j] - mu) * inv;
        xo[c] = f2bf((1.f + mod[HIDDEN + c]) * xn + mod[c]);
    }
}

// RMS-norm + RoPE, fp32 qkv -> head-major bf16 Q/K/V. one block per (pos, head).
__global__ __launch_bounds__(128)
void qkv_prep(const float* __restrict__ qkv, const float* __restrict__ qs,
              const float* __restrict__ ks, const float* __restrict__ pe,
              bf16* __restrict__ Q, bf16* __restrict__ K, bf16* __restrict__ V) {
    __shared__ float sq[128], sk[128], rq[128], rk[128];
    int pos = blockIdx.x, h = blockIdx.y, d = threadIdx.x;
    long base = (long)pos * (3*HIDDEN) + h * HDIM + d;
    float q = qkv[base], k = qkv[base + HIDDEN], v = qkv[base + 2*HIDDEN];
    rq[d] = q * q; rk[d] = k * k;
    __syncthreads();
    for (int s = 64; s > 0; s >>= 1) {
        if (d < s) { rq[d] += rq[d+s]; rk[d] += rk[d+s]; }
        __syncthreads();
    }
    float qn = q * rsqrtf(rq[0] / 128.f + 1e-6f) * qs[d];
    float kn = k * rsqrtf(rk[0] / 128.f + 1e-6f) * ks[d];
    sq[d] = qn; sk[d] = kn;
    __syncthreads();
    int p = d >> 1, j = d & 1;
    const float* peb = pe + ((long)pos * 64 + p) * 4 + j * 2;
    float qo = peb[0] * sq[2*p] + peb[1] * sq[2*p+1];
    float ko = peb[0] * sk[2*p] + peb[1] * sk[2*p+1];
    long ob = ((long)h * SEQ + pos) * HDIM + d;
    Q[ob] = f2bf(qo); K[ob] = f2bf(ko); V[ob] = f2bf(v);
}

// ---------------- tiled bf16 WMMA GEMM: C[M,N] = A[M,K] * Bt[N,K]^T ----------------
// 256x128 block tile, K step 32, 8 waves: wave w owns rows 32w..32w+31
// (2 A-frags, 16 WMMA/step sharing 8 B-frags). Both operands contiguous along K,
// so all LDS traffic is ds_load/store_b128.
// MODE 0: lin1 epilogue (qkv fp32 + gelu->bf16 cat).  MODE 1: lin2 epilogue (residual+gate).
template <int MODE>
__global__ __launch_bounds__(256)
void gemm_bf16(const bf16* __restrict__ A, int lda,
               const bf16* __restrict__ Bt, int ldbt, int K,
               const float* __restrict__ bias,
               float* __restrict__ qkv_out, bf16* __restrict__ cat_out,
               const float* __restrict__ xres, const float* __restrict__ gate,
               float* __restrict__ out) {
    __shared__ __align__(16) bf16 sA[256 * 32];    // [m][k]
    __shared__ __align__(16) bf16 sB[128 * 32];    // [n][k]
    const int t = threadIdx.x;
    const int wave = t >> 5, lane = t & 31;
    const int lh = lane & 15, lg = lane >> 4;
    const int n0 = blockIdx.x * 128, m0 = blockIdx.y * 256;

    f32x8 acc[2][8];
#pragma unroll
    for (int mi = 0; mi < 2; ++mi)
#pragma unroll
        for (int i = 0; i < 8; ++i) acc[mi][i] = (f32x8){0.f,0.f,0.f,0.f,0.f,0.f,0.f,0.f};

    for (int kb = 0; kb < K; kb += 32) {
#pragma unroll
        for (int i = 0; i < 4; ++i) {           // A tile: 1024 x 16B chunks
            int c = t + i * 256;
            int m = c >> 2, kc = c & 3;
            *(u32x4*)&sA[m*32 + kc*8] = *(const u32x4*)&A[(long)(m0+m)*lda + kb + kc*8];
        }
#pragma unroll
        for (int i = 0; i < 2; ++i) {           // B tile (pre-transposed weights)
            int c = t + i * 256;
            int n = c >> 2, kc = c & 3;
            *(u32x4*)&sB[n*32 + kc*8] = *(const u32x4*)&Bt[(long)(n0+n)*ldbt + kb + kc*8];
        }
        __syncthreads();

        FragB a0, a1;
        int row0 = wave*32 + lh;
        a0.q[0] = *(const u32x4*)&sA[row0*32 + lg*8];
        a0.q[1] = *(const u32x4*)&sA[row0*32 + lg*8 + 16];
        a1.q[0] = *(const u32x4*)&sA[(row0+16)*32 + lg*8];
        a1.q[1] = *(const u32x4*)&sA[(row0+16)*32 + lg*8 + 16];
#pragma unroll
        for (int nt = 0; nt < 8; ++nt) {
            FragB b;
            int ncol = nt*16 + lh;
            b.q[0] = *(const u32x4*)&sB[ncol*32 + lg*8];
            b.q[1] = *(const u32x4*)&sB[ncol*32 + lg*8 + 16];
            acc[0][nt] = __builtin_amdgcn_wmma_f32_16x16x32_bf16(
                false, a0.v, false, b.v, (short)0, acc[0][nt], false, false);
            acc[1][nt] = __builtin_amdgcn_wmma_f32_16x16x32_bf16(
                false, a1.v, false, b.v, (short)0, acc[1][nt], false, false);
        }
        __syncthreads();
    }

#pragma unroll
    for (int mi = 0; mi < 2; ++mi) {
#pragma unroll
        for (int nt = 0; nt < 8; ++nt) {
            int gc = n0 + nt*16 + lh;
            float bv = bias[gc];
#pragma unroll
            for (int v = 0; v < 8; ++v) {
                int gr = m0 + wave*32 + mi*16 + v + 8*lg;   // WMMA D row mapping
                float val = acc[mi][nt][v] + bv;
                if (MODE == 0) {
                    if (gc < 3*HIDDEN) {
                        qkv_out[(long)gr * (3*HIDDEN) + gc] = val;
                    } else {
                        float xg = val;
                        float g = 0.5f * xg * (1.f + tanhf(0.7978845608028654f *
                                       (xg + 0.044715f * xg * xg * xg)));
                        cat_out[(long)gr * CATW + (gc - 3*HIDDEN) + HIDDEN] = f2bf(g);
                    }
                } else {
                    out[(long)gr * HIDDEN + gc] =
                        xres[(long)gr * HIDDEN + gc] + gate[gc] * val;
                }
            }
        }
    }
}

// ---------------- flash attention, bf16 WMMA, online softmax ----------------
// block: 1 head, 128 Q rows (wave w owns rows 16w..16w+15). KV tile = 32.
// V comes in pre-transposed [h][d][pos] so all staging is b128 copies.
__global__ __launch_bounds__(256)
void attn_kernel(const bf16* __restrict__ Qm, const bf16* __restrict__ Km,
                 const bf16* __restrict__ Vt, bf16* __restrict__ cat_out) {
    __shared__ __align__(16) bf16 sK[32 * 128];    // [pos][d]
    __shared__ __align__(16) bf16 sVt[128 * 32];   // [d][pos]
    __shared__ __align__(16) bf16 sP[8][16 * 32];  // per-wave P tile [m][kv]

    const int t = threadIdx.x, wave = t >> 5, lane = t & 31;
    const int lh = lane & 15, lg = lane >> 4;
    const int head = blockIdx.y;
    const int q0 = blockIdx.x * 128 + wave * 16;

    const bf16* qh  = Qm + (long)head * SEQ * HDIM;
    const bf16* kh  = Km + (long)head * SEQ * HDIM;
    const bf16* vth = Vt + (long)head * SEQ * HDIM;   // [128][2048]

    FragB qf[4];
#pragma unroll
    for (int kc = 0; kc < 4; ++kc) {
        const bf16* p = qh + (long)(q0 + lh) * HDIM + kc*32 + lg*8;
        qf[kc].q[0] = *(const u32x4*)p;
        qf[kc].q[1] = *(const u32x4*)(p + 16);
    }

    f32x8 o[8];
#pragma unroll
    for (int i = 0; i < 8; ++i) o[i] = (f32x8){0.f,0.f,0.f,0.f,0.f,0.f,0.f,0.f};
    float m_i[8], l_i[8];
#pragma unroll
    for (int v = 0; v < 8; ++v) { m_i[v] = -1e30f; l_i[v] = 0.f; }

    const float scale = 0.08838834764831845f;   // 1/sqrt(128)

    for (int j0 = 0; j0 < SEQ; j0 += 32) {
#pragma unroll
        for (int i = 0; i < 2; ++i) {           // K rows: b128 copies
            int c = t + i * 256;
            int pos = c >> 4, dc = c & 15;
            *(u32x4*)&sK[pos*128 + dc*8] = *(const u32x4*)&kh[(long)(j0+pos)*HDIM + dc*8];
        }
#pragma unroll
        for (int i = 0; i < 2; ++i) {           // V columns (pre-transposed): b128 copies
            int c = t + i * 256;
            int dd = c >> 2, pc = c & 3;
            *(u32x4*)&sVt[dd*32 + pc*8] = *(const u32x4*)&vth[(long)dd*SEQ + j0 + pc*8];
        }
        __syncthreads();

        // S = Q * K^T : two 16x16 tiles (kv cols 0..15 / 16..31)
        f32x8 s0 = (f32x8){0.f,0.f,0.f,0.f,0.f,0.f,0.f,0.f};
        f32x8 s1 = s0;
#pragma unroll
        for (int kc = 0; kc < 4; ++kc) {
            FragB b0, b1;
            const bf16* r0 = &sK[(lh      ) * 128 + kc*32 + lg*8];
            const bf16* r1 = &sK[(lh + 16 ) * 128 + kc*32 + lg*8];
            b0.q[0] = *(const u32x4*)r0; b0.q[1] = *(const u32x4*)(r0 + 16);
            b1.q[0] = *(const u32x4*)r1; b1.q[1] = *(const u32x4*)(r1 + 16);
            s0 = __builtin_amdgcn_wmma_f32_16x16x32_bf16(false, qf[kc].v, false, b0.v, (short)0, s0, false, false);
            s1 = __builtin_amdgcn_wmma_f32_16x16x32_bf16(false, qf[kc].v, false, b1.v, (short)0, s1, false, false);
        }

        float corr[8];
#pragma unroll
        for (int v = 0; v < 8; ++v) {
            float a0 = s0[v] * scale, a1 = s1[v] * scale;
            float mx = fmaxf(a0, a1);
#pragma unroll
            for (int off = 1; off < 16; off <<= 1)
                mx = fmaxf(mx, __shfl_xor(mx, off, 16));
            float mn = fmaxf(m_i[v], mx);
            corr[v] = __expf(m_i[v] - mn);
            float p0 = __expf(a0 - mn), p1 = __expf(a1 - mn);
            int row = v + 8*lg;
            sP[wave][row*32 + lh]      = f2bf(p0);
            sP[wave][row*32 + lh + 16] = f2bf(p1);
            float rs = p0 + p1;
#pragma unroll
            for (int off = 1; off < 16; off <<= 1)
                rs += __shfl_xor(rs, off, 16);
            l_i[v] = l_i[v] * corr[v] + rs;
            m_i[v] = mn;
        }
#pragma unroll
        for (int nt = 0; nt < 8; ++nt)
#pragma unroll
            for (int v = 0; v < 8; ++v) o[nt][v] *= corr[v];

        // re-shape P into A-fragment layout via per-wave LDS (in-order DS per wave)
        FragB pf;
        {
            const bf16* p = &sP[wave][lh*32 + lg*8];
            pf.q[0] = *(const u32x4*)p;
            pf.q[1] = *(const u32x4*)(p + 16);
        }
#pragma unroll
        for (int nt = 0; nt < 8; ++nt) {
            FragB b;
            const bf16* p = &sVt[(nt*16 + lh)*32 + lg*8];
            b.q[0] = *(const u32x4*)p; b.q[1] = *(const u32x4*)(p + 16);
            o[nt] = __builtin_amdgcn_wmma_f32_16x16x32_bf16(false, pf.v, false, b.v, (short)0, o[nt], false, false);
        }
        __syncthreads();
    }

#pragma unroll
    for (int v = 0; v < 8; ++v) l_i[v] = 1.0f / l_i[v];
#pragma unroll
    for (int nt = 0; nt < 8; ++nt) {
        int col = head * HDIM + nt*16 + lh;
#pragma unroll
        for (int v = 0; v < 8; ++v) {
            int row = q0 + v + 8*lg;
            cat_out[(long)row * CATW + col] = f2bf(o[nt][v] * l_i[v]);
        }
    }
}

// ---------------- host launch ----------------
extern "C" void kernel_launch(void* const* d_in, const int* in_sizes, int n_in,
                              void* d_out, int out_size, void* d_ws, size_t ws_size,
                              hipStream_t stream) {
    (void)in_sizes; (void)n_in; (void)out_size; (void)ws_size;
    const float* x       = (const float*)d_in[0];
    const float* vec     = (const float*)d_in[1];
    const float* pe      = (const float*)d_in[2];
    const float* mod_w   = (const float*)d_in[3];
    const float* mod_b   = (const float*)d_in[4];
    const float* lin1_w  = (const float*)d_in[5];
    const float* lin1_b  = (const float*)d_in[6];
    const float* lin2_w  = (const float*)d_in[7];
    const float* lin2_b  = (const float*)d_in[8];
    const float* q_scale = (const float*)d_in[9];
    const float* k_scale = (const float*)d_in[10];
    float* out = (float*)d_out;

    char* ws = (char*)d_ws;
    size_t off = 0;
    auto alloc = [&](size_t bytes) { void* p = ws + off; off = (off + bytes + 255) & ~(size_t)255; return p; };
    float* w_mod  = (float*)alloc(9216 * 4);
    float* w_sv   = (float*)alloc(3072 * 4);
    bf16*  w_xmod = (bf16*) alloc((size_t)SEQ * HIDDEN * 2);
    bf16*  w_w1t  = (bf16*) alloc((size_t)N1 * HIDDEN * 2);     // [21504][3072]
    bf16*  w_w2t  = (bf16*) alloc((size_t)HIDDEN * CATW * 2);   // [3072][15360]
    float* w_qkv  = (float*)alloc((size_t)SEQ * 3 * HIDDEN * 4);
    bf16*  w_q    = (bf16*) alloc((size_t)NHEADS * SEQ * HDIM * 2);
    bf16*  w_k    = (bf16*) alloc((size_t)NHEADS * SEQ * HDIM * 2);
    bf16*  w_v    = (bf16*) alloc((size_t)NHEADS * SEQ * HDIM * 2);
    bf16*  w_vt   = (bf16*) alloc((size_t)NHEADS * SEQ * HDIM * 2);
    bf16*  w_cat  = (bf16*) alloc((size_t)SEQ * CATW * 2);

    // 1. modulation path
    silu_k<<<(HIDDEN + 255) / 256, 256, 0, stream>>>(vec, w_sv);
    mod_gemv<<<(3*HIDDEN) / 256, 256, 0, stream>>>(w_sv, mod_w, mod_b, w_mod);

    // 2. weight downconversion + transpose (bf16 copies fit in L2 for the GEMM passes)
    cast_transpose<<<dim3(HIDDEN/32, N1/32), 256, 0, stream>>>(lin1_w, w_w1t, HIDDEN, N1);
    cast_transpose<<<dim3(CATW/32, HIDDEN/32), 256, 0, stream>>>(lin2_w, w_w2t, CATW, HIDDEN);

    // 3. LayerNorm + modulate -> bf16
    ln_mod<<<SEQ, 256, 0, stream>>>(x, w_mod, w_xmod);

    // 4. lin1 GEMM (2048 x 21504, K=3072) with fused split/gelu epilogue
    gemm_bf16<0><<<dim3(N1 / 128, SEQ / 256), 256, 0, stream>>>(
        w_xmod, HIDDEN, w_w1t, HIDDEN, HIDDEN, lin1_b,
        w_qkv, w_cat, nullptr, nullptr, nullptr);

    // 5. RMS-norm + RoPE -> head-major bf16 Q/K/V, then V transpose for attention
    qkv_prep<<<dim3(SEQ, NHEADS), 128, 0, stream>>>(
        w_qkv, q_scale, k_scale, pe, w_q, w_k, w_v);
    transpose_v<<<dim3(SEQ/32, HDIM/32, NHEADS), 256, 0, stream>>>(w_v, w_vt);

    // 6. flash attention -> cat[:, 0:3072]
    attn_kernel<<<dim3(SEQ / 128, NHEADS), 256, 0, stream>>>(w_q, w_k, w_vt, w_cat);

    // 7. lin2 GEMM (2048 x 3072, K=15360) with fused residual+gate epilogue
    gemm_bf16<1><<<dim3(HIDDEN / 128, SEQ / 256), 256, 0, stream>>>(
        w_cat, CATW, w_w2t, CATW, CATW, lin2_b,
        nullptr, nullptr, x, w_mod + 2*HIDDEN, out);
}
